// LMClassifier_27023934226772
// MI455X (gfx1250) — matrix operand order repaired and checked by reading, not compiled, not run on hardware
//
#include <hip/hip_runtime.h>
#include <hip/hip_bf16.h>

#define TDIM 256
#define BDIM 32
#define HDIM 2048
#define EDIM 1024
#define VDIM 50257
#define NROWS ((TDIM - 2) * BDIM)   /* 8128 */
#define VPAD  50432                 /* 197 * 256 */
#define NTILE (VPAD / 256)          /* 197 */

typedef __attribute__((ext_vector_type(16))) __bf16       v16bf;
typedef __attribute__((ext_vector_type(8)))  float        v8f;
typedef __attribute__((ext_vector_type(4)))  unsigned int u32x4;

union FragU { v16bf v; u32x4 q[2]; };

__device__ __forceinline__ unsigned short f2bf(float x) {
  unsigned int u = __float_as_uint(x);
  u += 0x7fffu + ((u >> 16) & 1u);       // round-to-nearest-even
  return (unsigned short)(u >> 16);
}

// A fragment (16x32 bf16): lane = row (lane%16); half=lane/16 selects K slices
// [half*8, half*8+8) -> VGPR0..3 and [16+half*8, +8) -> VGPR4..7  (ISA 7.12.2)
__device__ __forceinline__ v16bf load_a_frag(const unsigned short* A, int lda,
                                             int k, int lane) {
  const int m    = lane & 15;
  const int half = lane >> 4;
  const unsigned short* p = A + (size_t)m * lda + (k + half * 8);
  FragU f;
  f.q[0] = *(const u32x4*)p;
  f.q[1] = *(const u32x4*)(p + 16);
  return f.v;
}

// B fragment (32x16 bf16): lane = col (lane%16); half=lane/16 selects K slices
// [half*8, +8) and [16+half*8, +8). B column n == row (col0+n) of row-major W.
__device__ __forceinline__ v16bf load_b_frag(const unsigned short* Wt, size_t ldb,
                                             int col0, int k, int lane) {
  const int n    = lane & 15;
  const int half = lane >> 4;
  const unsigned short* p = Wt + (size_t)(col0 + n) * ldb + (k + half * 8);
  FragU f;
  f.q[0] = *(const u32x4*)p;
  f.q[1] = *(const u32x4*)(p + 16);
  return f.v;
}

// ---------------- conversion / setup kernels ----------------

__global__ __launch_bounds__(256) void build_ctx(const float* __restrict__ hidden,
                                                 unsigned short* __restrict__ ctx) {
  const int nrow = blockIdx.x;            // 0..NROWS-1 ; n = t*B + b
  const int t = nrow >> 5;
  const int b = nrow & 31;
  const int c0 = threadIdx.x * 8;         // 2048/256 = 8 cols per thread
  const int tt = (c0 < (HDIM / 2)) ? t : (t + 2);
  const float* src = hidden + ((size_t)tt * BDIM + b) * HDIM + c0;
  unsigned short* dst = ctx + (size_t)nrow * HDIM + c0;
#pragma unroll
  for (int j = 0; j < 8; ++j) dst[j] = f2bf(src[j]);
}

__global__ __launch_bounds__(256) void conv_bf16(const float* __restrict__ src,
                                                 unsigned short* __restrict__ dst,
                                                 long nquad) {
  long i = (long)blockIdx.x * 256 + threadIdx.x;
  if (i < nquad) {
#pragma unroll
    for (int j = 0; j < 4; ++j) dst[i * 4 + j] = f2bf(src[i * 4 + j]);
  }
}

__global__ __launch_bounds__(256) void conv_w2_pad(const float* __restrict__ w2,
                                                   unsigned short* __restrict__ w2b) {
  const int row = blockIdx.x;             // 0..VPAD-1
  const int c0  = threadIdx.x * 4;        // 1024/256 = 4 cols per thread
  unsigned short* dst = w2b + (size_t)row * EDIM + c0;
  if (row < VDIM) {
    const float* src = w2 + (size_t)row * EDIM + c0;
#pragma unroll
    for (int j = 0; j < 4; ++j) dst[j] = f2bf(src[j]);
  } else {
#pragma unroll
    for (int j = 0; j < 4; ++j) dst[j] = 0;
  }
}

__global__ __launch_bounds__(256) void pad_b2_init(const float* __restrict__ b2,
                                                   float* __restrict__ b2p,
                                                   float* __restrict__ accum) {
  const int i = blockIdx.x * 256 + threadIdx.x;
  if (i < VPAD) b2p[i] = (i < VDIM) ? b2[i] : -1e30f;  // pad cols vanish in softmax
  if (i == 0) { accum[0] = 0.0f; accum[1] = 0.0f; }
}

// ---------------- GEMM1: emb = sigmoid(ctx @ W1^T + b1) ----------------
// grid (NROWS/16, EDIM/128); block 256 = 8 waves; wave -> one 16x16 tile

__global__ __launch_bounds__(256) void gemm1_sigmoid(
    const unsigned short* __restrict__ ctx,   // NROWS x HDIM (bf16)
    const unsigned short* __restrict__ w1b,   // EDIM x HDIM (bf16)
    const float* __restrict__ b1,
    unsigned short* __restrict__ emb)         // NROWS x EDIM (bf16)
{
  const int lane = threadIdx.x & 31;
  const int wave = threadIdx.x >> 5;
  const int row0 = blockIdx.x * 16;
  const int col0 = blockIdx.y * 128 + wave * 16;
  const unsigned short* Arow = ctx + (size_t)row0 * HDIM;

  v8f acc = {};
  for (int k = 0; k < HDIM; k += 32) {
    v16bf a = load_a_frag(Arow, HDIM, k, lane);
    v16bf b = load_b_frag(w1b, HDIM, col0, k, lane);
    acc = __builtin_amdgcn_wmma_f32_16x16x32_bf16(false, a, false, b,
                                                  (short)0, acc, false, false);
  }
  const int n  = lane & 15;
  const int rb = (lane >> 4) * 8;
  const float bias = b1[col0 + n];
  unsigned short* outp = emb + (size_t)(row0 + rb) * EDIM + (col0 + n);
#pragma unroll
  for (int i = 0; i < 8; ++i) {
    float x = acc[i] + bias;
    float s = 1.0f / (1.0f + __expf(-x));
    outp[(size_t)i * EDIM] = f2bf(s);
  }
}

// ---------------- GEMM2 + streaming log-softmax + masked NLL ----------------
// grid (NROWS/32) = 254 blocks; block owns 32 rows (= one timestep t, all b),
// sweeps VPAD in 256-col tiles. Each wave: 32x32 output = 4 WMMAs per k-step
// (2 A-frags x 2 B-frags) -> 2x reuse on both operands vs 16-row blocking.
// A (32x1024 bf16 = 64KB) read from global; stays hot in WGP$/L2.

__global__ __launch_bounds__(256) void gemm2_loss(
    const unsigned short* __restrict__ emb,   // NROWS x EDIM (bf16)
    const unsigned short* __restrict__ w2b,   // VPAD  x EDIM (bf16)
    const float* __restrict__ b2p,            // VPAD
    const int* __restrict__ token,            // T x B
    const int* __restrict__ lens,             // B
    const float* __restrict__ inv_temp,
    float* __restrict__ accum)                // [sum(nll*mask), sum(mask)]
{
  __shared__ float sLog[32 * 256];            // 32 KB logits tile
  __shared__ float sPm[32][8];
  __shared__ float sPs[32][8];
  __shared__ float sM[32], sS[32], sTl[32], sMask[32];
  __shared__ float sNll[32], sCnt[32];
  __shared__ int   sTgt[32];

  const int tid  = threadIdx.x;
  const int lane = tid & 31;
  const int wave = tid >> 5;
  const int row0 = blockIdx.x * 32;           // == t * 32
  const int t    = blockIdx.x;

  if (tid < 32) {
    const int b = tid;
    sTgt[b]  = token[(t + 1) * BDIM + b];
    sMask[b] = (t < lens[b] - 2) ? 1.0f : 0.0f;
    sM[b] = -1e30f; sS[b] = 0.0f; sTl[b] = 0.0f;
  }
  __syncthreads();

  const float it = inv_temp[0];
  const int n  = lane & 15;
  const int rb = (lane >> 4) * 8;
  const unsigned short* A0 = emb + (size_t)row0 * EDIM;        // rows 0..15
  const unsigned short* A1 = emb + (size_t)(row0 + 16) * EDIM; // rows 16..31

  for (int vt = 0; vt < NTILE; ++vt) {
    const int tilebase = vt * 256;
    const int col0 = tilebase + wave * 32;

    if (vt + 1 < NTILE) {  // prefetch next tile's B rows for this wave
      __builtin_prefetch(w2b + (size_t)(col0 + 256 + n) * EDIM, 0, 1);
      __builtin_prefetch(w2b + (size_t)(col0 + 272 + n) * EDIM, 0, 1);
    }

    v8f acc00 = {}, acc01 = {}, acc10 = {}, acc11 = {};
    for (int k = 0; k < EDIM; k += 32) {
      v16bf a0  = load_a_frag(A0, EDIM, k, lane);
      v16bf a1  = load_a_frag(A1, EDIM, k, lane);
      v16bf bf0 = load_b_frag(w2b, EDIM, col0,      k, lane);
      v16bf bf1 = load_b_frag(w2b, EDIM, col0 + 16, k, lane);
      acc00 = __builtin_amdgcn_wmma_f32_16x16x32_bf16(false, a0, false, bf0,
                                                      (short)0, acc00, false, false);
      acc01 = __builtin_amdgcn_wmma_f32_16x16x32_bf16(false, a0, false, bf1,
                                                      (short)0, acc01, false, false);
      acc10 = __builtin_amdgcn_wmma_f32_16x16x32_bf16(false, a1, false, bf0,
                                                      (short)0, acc10, false, false);
      acc11 = __builtin_amdgcn_wmma_f32_16x16x32_bf16(false, a1, false, bf1,
                                                      (short)0, acc11, false, false);
    }
    const float bias0 = b2p[col0 + n];
    const float bias1 = b2p[col0 + 16 + n];
#pragma unroll
    for (int i = 0; i < 8; ++i) {
      sLog[(rb + i) * 256 + (wave * 32 + n)]           = (acc00[i] + bias0) * it;
      sLog[(rb + i) * 256 + (wave * 32 + 16 + n)]      = (acc01[i] + bias1) * it;
      sLog[(16 + rb + i) * 256 + (wave * 32 + n)]      = (acc10[i] + bias0) * it;
      sLog[(16 + rb + i) * 256 + (wave * 32 + 16 + n)] = (acc11[i] + bias1) * it;
    }
    __syncthreads();

    {   // per-row partial max/sumexp: 8 threads per row, 32 cols each
      const int r = tid >> 3;
      const int c = tid & 7;
      const float* rowp = &sLog[r * 256 + c * 32];
      float pm = -1e30f;
#pragma unroll
      for (int j = 0; j < 32; ++j) pm = fmaxf(pm, rowp[j]);
      float ps = 0.0f;
#pragma unroll
      for (int j = 0; j < 32; ++j) ps += __expf(rowp[j] - pm);
      sPm[r][c] = pm;
      sPs[r][c] = ps;
    }
    __syncthreads();

    if (tid < 32) {  // row leader: online-softmax merge + target capture
      const int r = tid;
      float m = sM[r], s = sS[r];
#pragma unroll
      for (int c = 0; c < 8; ++c) {
        const float pm = sPm[r][c], ps = sPs[r][c];
        const float nm = fmaxf(m, pm);
        s = s * __expf(m - nm) + ps * __expf(pm - nm);
        m = nm;
      }
      sM[r] = m; sS[r] = s;
      const int tc = sTgt[r] - tilebase;
      if (tc >= 0 && tc < 256) sTl[r] = sLog[r * 256 + tc];
    }
    __syncthreads();
  }

  if (tid < 32) {
    const float nll = sM[tid] + __logf(sS[tid]) - sTl[tid];
    sNll[tid] = (sMask[tid] > 0.0f) ? nll : 0.0f;
    sCnt[tid] = sMask[tid];
  }
  __syncthreads();
  if (tid == 0) {
    float a = 0.0f, c = 0.0f;
#pragma unroll
    for (int r = 0; r < 32; ++r) { a += sNll[r]; c += sCnt[r]; }
    atomicAdd(&accum[0], a);
    atomicAdd(&accum[1], c);
  }
}

__global__ void finalize_k(const float* __restrict__ accum, float* __restrict__ out) {
  if (threadIdx.x == 0 && blockIdx.x == 0) out[0] = accum[0] / accum[1];
}

// ---------------- host entry ----------------

extern "C" void kernel_launch(void* const* d_in, const int* in_sizes, int n_in,
                              void* d_out, int out_size, void* d_ws, size_t ws_size,
                              hipStream_t stream) {
  (void)in_sizes; (void)n_in; (void)out_size; (void)ws_size;
  const float* hidden   = (const float*)d_in[0];
  const int*   lens     = (const int*)  d_in[1];
  const int*   token    = (const int*)  d_in[2];
  const float* W1       = (const float*)d_in[3];
  const float* b1       = (const float*)d_in[4];
  const float* W2       = (const float*)d_in[5];
  const float* b2       = (const float*)d_in[6];
  const float* inv_temp = (const float*)d_in[7];
  float* out = (float*)d_out;

  char* ws = (char*)d_ws;
  size_t off = 0;
  auto wsalloc = [&](size_t bytes) {
    void* p = ws + off;
    off = (off + bytes + 255) & ~(size_t)255;
    return p;
  };
  unsigned short* ctx  = (unsigned short*)wsalloc((size_t)NROWS * HDIM * 2); // 33.3 MB
  unsigned short* w1b  = (unsigned short*)wsalloc((size_t)EDIM  * HDIM * 2); //  4.0 MB
  unsigned short* w2b  = (unsigned short*)wsalloc((size_t)VPAD  * EDIM * 2); // 103 MB (fits L2)
  float*          b2p  = (float*)         wsalloc((size_t)VPAD * 4);
  unsigned short* embp = (unsigned short*)wsalloc((size_t)NROWS * EDIM * 2); // 16.6 MB
  float*          accum= (float*)         wsalloc(256);

  hipLaunchKernelGGL(build_ctx, dim3(NROWS), dim3(256), 0, stream, hidden, ctx);
  hipLaunchKernelGGL(conv_bf16, dim3((EDIM * (long)HDIM / 4 + 255) / 256), dim3(256), 0,
                     stream, W1, w1b, (long)(EDIM * (long)HDIM / 4));
  hipLaunchKernelGGL(conv_w2_pad, dim3(VPAD), dim3(256), 0, stream, W2, w2b);
  hipLaunchKernelGGL(pad_b2_init, dim3(VPAD / 256), dim3(256), 0, stream, b2, b2p, accum);
  hipLaunchKernelGGL(gemm1_sigmoid, dim3(NROWS / 16, EDIM / 128), dim3(256), 0, stream,
                     ctx, w1b, b1, embp);
  hipLaunchKernelGGL(gemm2_loss, dim3(NROWS / 32), dim3(256), 0, stream,
                     embp, w2b, b2p, token, lens, inv_temp, accum);
  hipLaunchKernelGGL(finalize_k, dim3(1), dim3(1), 0, stream, accum, out);
}
